// EdgeModel_12077448036505
// MI455X (gfx1250) — matrix operand order repaired and checked by reading, not compiled, not run on hardware
//
#include <hip/hip_runtime.h>

typedef __bf16 v16bf __attribute__((ext_vector_type(16)));
typedef __bf16 v8bf  __attribute__((ext_vector_type(8)));
typedef __bf16 v4bf  __attribute__((ext_vector_type(4)));
typedef float  v8f   __attribute__((ext_vector_type(8)));
typedef float  v4f   __attribute__((ext_vector_type(4)));

#define E_TOTAL   500000
#define NODE_DIM  128
#define EDGE_DIM  64
#define U_DIM     64
#define IN_DIM    384        // 2*128 + 64 + 64
#define OUT_DIM   128
#define M_TILE    64         // edges per workgroup
#define KSTEPS    12         // 384 / 32
#define LDS_STRIDE 392       // 384 + 8 bf16 pad -> 784B row stride (breaks bank periodicity)

// ---- pre-kernel: convert W [128x384] fp32 -> bf16 into workspace ----
__global__ void wmlp_convert_w(const float* __restrict__ W,
                               __bf16* __restrict__ Wb) {
  int i = (blockIdx.x * blockDim.x + threadIdx.x) * 4;   // 48*256*4 == 49152 exactly
  v4f w = *(const v4f*)(W + i);
  v4bf o;
  o[0] = (__bf16)w[0]; o[1] = (__bf16)w[1];
  o[2] = (__bf16)w[2]; o[3] = (__bf16)w[3];
  *(v4bf*)(Wb + i) = o;
}

// ---- main fused gather+concat+GEMM+bias+ReLU kernel ----
__global__ __launch_bounds__(256)
void wmlp_edge_mlp(const float* __restrict__ src,
                   const float* __restrict__ dst,
                   const float* __restrict__ edge,
                   const float* __restrict__ u,
                   const int*   __restrict__ batch,
                   const __bf16* __restrict__ Wb,
                   const float* __restrict__ bias,
                   float* __restrict__ out) {
  __shared__ __bf16 xs[M_TILE * LDS_STRIDE];   // 64 x (384+8) bf16 = 50176 B

  const int tid  = threadIdx.x;
  const int lane = tid & 31;
  const int wave = tid >> 5;          // 8 waves: one 16-wide N tile each
  const int e0   = blockIdx.x * M_TILE;

  const int n  = lane & 15;           // N (output channel) within tile / M row for A
  const int h  = lane >> 4;           // lane half selects K sub-chunk
  const int o0 = wave * 16;

  // --- B fragments: W[o][k] is already the B-operand layout (N-major rows).
  // Per lane: N = o0+n, K contiguous [kb*32 + h*16, +16), packed 2/VGPR.
  v16bf bfrag[KSTEPS];
  {
    const __bf16* wrow = Wb + (size_t)(o0 + n) * IN_DIM + h * 16;
#pragma unroll
    for (int kb = 0; kb < KSTEPS; ++kb) {
      v8bf lo = *(const v8bf*)(wrow + kb * 32);
      v8bf hi = *(const v8bf*)(wrow + kb * 32 + 8);
      bfrag[kb] = __builtin_shufflevector(lo, hi,
          0, 1, 2, 3, 4, 5, 6, 7, 8, 9, 10, 11, 12, 13, 14, 15);
    }
  }
  const float bval = bias[o0 + n];

  // --- cooperative stage: gather + concat + fp32->bf16 into LDS.
  // 4 threads per row, each covers 24 float4 chunks (96 floats).
  {
    const int row = tid >> 2;         // 0..63
    const int q   = tid & 3;
    const int e   = e0 + row;
    if (e < E_TOTAL) {
      const int bidx = batch[e];
      const float* s0 = src  + (size_t)e    * NODE_DIM;
      const float* s1 = dst  + (size_t)e    * NODE_DIM;
      const float* s2 = edge + (size_t)e    * EDGE_DIM;
      const float* s3 = u    + (size_t)bidx * U_DIM;
      __bf16* ld = xs + row * LDS_STRIDE;
#pragma unroll
      for (int j = 0; j < 24; ++j) {
        const int c = (q * 24 + j) * 4;   // region edges (128/256/320) are 4-aligned
        const float* p;
        if (c < 128)      p = s0 + c;
        else if (c < 256) p = s1 + (c - 128);
        else if (c < 320) p = s2 + (c - 256);
        else              p = s3 + (c - 320);
        v4f v = *(const v4f*)p;
        v4bf o;
        o[0] = (__bf16)v[0]; o[1] = (__bf16)v[1];
        o[2] = (__bf16)v[2]; o[3] = (__bf16)v[3];
        *(v4bf*)(ld + c) = o;
      }
    }
  }
  __syncthreads();

  // --- WMMA over 16-row subtiles (E is a multiple of 16: exact guard, no row masks)
  const int rem    = (E_TOTAL - e0) >> 4;
  const int stiles = rem < 4 ? rem : 4;
  for (int s = 0; s < stiles; ++s) {
    v8f acc = {};
    // A fragment as loaded here: lane m = n, K contiguous [kb*32 + h*16, +16), same k order as B
    const __bf16* arow = xs + (s * 16 + n) * LDS_STRIDE + h * 16;
#pragma unroll
    for (int kb = 0; kb < KSTEPS; ++kb) {
      v8bf lo = *(const v8bf*)(arow + kb * 32);
      v8bf hi = *(const v8bf*)(arow + kb * 32 + 8);
      v16bf afrag = __builtin_shufflevector(lo, hi,
          0, 1, 2, 3, 4, 5, 6, 7, 8, 9, 10, 11, 12, 13, 14, 15);
      acc = __builtin_amdgcn_wmma_f32_16x16x32_bf16(
          /*neg_a=*/false, afrag, /*neg_b=*/false, bfrag[kb],
          /*c_mod=*/(short)0, acc, /*reuse_a=*/false, /*reuse_b=*/false);
    }
    // epilogue: C/D layout -> lanes 0-15: M=r, N=lane; lanes 16-31: M=8+r, N=lane-16
    const size_t m0 = (size_t)(e0 + s * 16 + h * 8);
    float* orow = out + m0 * OUT_DIM + o0 + n;
#pragma unroll
    for (int r = 0; r < 8; ++r) {
      float vv = acc[r] + bval;
      orow[(size_t)r * OUT_DIM] = vv > 0.0f ? vv : 0.0f;
    }
  }
}

extern "C" void kernel_launch(void* const* d_in, const int* in_sizes, int n_in,
                              void* d_out, int out_size, void* d_ws, size_t ws_size,
                              hipStream_t stream) {
  const float* src   = (const float*)d_in[0];
  const float* dst   = (const float*)d_in[1];
  const float* edge  = (const float*)d_in[2];
  const float* u     = (const float*)d_in[3];
  const int*   batch = (const int*)d_in[4];
  const float* W     = (const float*)d_in[5];
  const float* bias  = (const float*)d_in[6];
  float* out = (float*)d_out;
  __bf16* Wb = (__bf16*)d_ws;   // 128*384*2 = 98304 B of scratch

  wmlp_convert_w<<<48, 256, 0, stream>>>(W, Wb);       // 48*256*4 == 128*384

  const int nblocks = (E_TOTAL + M_TILE - 1) / M_TILE; // 7813
  wmlp_edge_mlp<<<nblocks, 256, 0, stream>>>(src, dst, edge, u, batch, Wb, bias, out);
}
